// MEATransformer_12678743458468
// MI455X (gfx1250) — compile-verified, matrix operand order checked by reading, not run on hardware
//
#include <hip/hip_runtime.h>
#include <hip/hip_bf16.h>

// ---------------- problem constants ----------------
#define NQ       256        // B: number of queries
#define NDB      500000     // N: database size (exact multiple of 32!)
#define DIM      256        // D
#define TOPK     3          // K
#define NLAB     12
#define NLAYERS_ 2
#define NSEQ     (NQ * TOPK)        // 768
#define MROWS    (NSEQ * 3)         // 2304 rows of H
#define SIMS_GRID 1024
#define NSTRIPS  (NDB / 32)         // 15625 (exact)
#define TPITCH   264                // bf16 pitch for transposed tile rows (528B, 16B-aligned)

typedef __bf16 bf16_t;
typedef bf16_t bf16x16 __attribute__((ext_vector_type(16)));
typedef bf16_t bf16x8  __attribute__((ext_vector_type(8)));
typedef float  f32x8   __attribute__((ext_vector_type(8)));

// Build a 16-element bf16 fragment from two contiguous 16-byte chunks.
// Pure SSA (shufflevector) -- no stack object, no scratch spills.
__device__ __forceinline__ bf16x16 make_frag(const bf16_t* p0, const bf16_t* p1) {
  bf16x8 lo = *(const bf16x8*)p0;
  bf16x8 hi = *(const bf16x8*)p1;
  return __builtin_shufflevector(lo, hi, 0, 1, 2, 3, 4, 5, 6, 7,
                                         8, 9, 10, 11, 12, 13, 14, 15);
}

__device__ __forceinline__ f32x8 wmma_bf16(bf16x16 a, bf16x16 b, f32x8 c) {
  return __builtin_amdgcn_wmma_f32_16x16x32_bf16(false, a, false, b, (short)0, c, false, false);
}

__device__ __forceinline__ void top3_update(float v, int c,
                                            float& tv0, float& tv1, float& tv2,
                                            int& ti0, int& ti1, int& ti2) {
  if (v > tv2) {
    if (v > tv0)      { tv2 = tv1; ti2 = ti1; tv1 = tv0; ti1 = ti0; tv0 = v; ti0 = c; }
    else if (v > tv1) { tv2 = tv1; ti2 = ti1; tv1 = v;  ti1 = c; }
    else              { tv2 = v;  ti2 = c; }
  }
}

// ---------------- 1) normalize queries -> bf16 ----------------
__global__ void normalize_kernel(const float* __restrict__ q, bf16_t* __restrict__ qn) {
  __shared__ float red[256];
  const int b = blockIdx.x, t = threadIdx.x;
  float v = q[b * DIM + t];
  red[t] = v * v;
  __syncthreads();
  for (int s = 128; s > 0; s >>= 1) {
    if (t < s) red[t] += red[t + s];
    __syncthreads();
  }
  float inv = rsqrtf(red[0]);
  qn[b * DIM + t] = (bf16_t)(v * inv);
}

// ---------------- fp32 -> bf16 convert ----------------
__global__ void cvt_kernel(const float* __restrict__ src, bf16_t* __restrict__ dst, int n) {
  int i = blockIdx.x * blockDim.x + threadIdx.x;
  if (i < n) dst[i] = (bf16_t)src[i];
}

// ---------------- 2) fused sims GEMM + per-strip top-3 ----------------
// 256 threads = 8 waves. Strip = 32 db columns x all 256 query rows, K=256.
// Full 256(K)x32(N) fp32 weight tile prefetched to registers (8 x b128/thread),
// double-buffered across strips; converted+transposed to LDS as bf16 once per
// strip; then 8 K-steps x 4 WMMA per wave with zero intervening barriers.
__global__ __launch_bounds__(256, 1)
void sims_topk_kernel(const bf16_t* __restrict__ qn,
                      const float*  __restrict__ weight,  // (D, N) row-major
                      float* __restrict__ partVal,
                      int*   __restrict__ partIdx) {
  __shared__ __align__(16) bf16_t tileT[32 * TPITCH];  // [n][k] transposed: 16.5 KB
  __shared__ float simsBuf[256 * 33];                  // 33 KB
  const int t    = threadIdx.x;
  const int lane = t & 31, wave = t >> 5;
  const int half = lane >> 4, l16 = lane & 15;
  const int q4 = (t & 7) * 4;   // this thread's 4-column group within the strip
  const int k0 = t >> 3;        // this thread's base k row; k = k0 + 32*i

  float tv0 = -3.4e38f, tv1 = -3.4e38f, tv2 = -3.4e38f;
  int   ti0 = 0, ti1 = 0, ti2 = 0;

  float4 buf[8];
  {  // prologue: prefetch first strip
    const int c0 = blockIdx.x * 32;
    #pragma unroll
    for (int i = 0; i < 8; ++i)
      buf[i] = *(const float4*)(weight + (size_t)(k0 + 32 * i) * NDB + (c0 + q4));
  }

  for (int s = blockIdx.x; s < NSTRIPS; s += SIMS_GRID) {
    const int c0 = s * 32;

    // deposit prefetched tile into LDS (fp32 -> bf16, transposed [n][k])
    #pragma unroll
    for (int i = 0; i < 8; ++i) {
      int k = k0 + 32 * i;
      tileT[(q4 + 0) * TPITCH + k] = (bf16_t)buf[i].x;
      tileT[(q4 + 1) * TPITCH + k] = (bf16_t)buf[i].y;
      tileT[(q4 + 2) * TPITCH + k] = (bf16_t)buf[i].z;
      tileT[(q4 + 3) * TPITCH + k] = (bf16_t)buf[i].w;
    }
    __syncthreads();  // (1) tile visible

    // kick off loads for the next strip (land while WMMAs run)
    const int sn = s + SIMS_GRID;
    if (sn < NSTRIPS) {
      const int c0n = sn * 32;
      #pragma unroll
      for (int i = 0; i < 8; ++i)
        buf[i] = *(const float4*)(weight + (size_t)(k0 + 32 * i) * NDB + (c0n + q4));
    }

    // wave w owns rows [32w, 32w+32): 2 M-tiles x 2 N-tiles
    f32x8 acc00 = {}, acc01 = {}, acc10 = {}, acc11 = {};
    #pragma unroll
    for (int kb = 0; kb < DIM; kb += 32) {
      bf16x16 b0 = make_frag(&tileT[(0  + l16) * TPITCH + kb + half * 8],
                             &tileT[(0  + l16) * TPITCH + kb + 16 + half * 8]);
      bf16x16 b1 = make_frag(&tileT[(16 + l16) * TPITCH + kb + half * 8],
                             &tileT[(16 + l16) * TPITCH + kb + 16 + half * 8]);
      const bf16_t* a0p = qn + (size_t)(wave * 32 + l16) * DIM + kb + half * 8;
      const bf16_t* a1p = a0p + 16 * DIM;
      bf16x16 a0 = make_frag(a0p, a0p + 16);
      bf16x16 a1 = make_frag(a1p, a1p + 16);
      acc00 = wmma_bf16(a0, b0, acc00);
      acc01 = wmma_bf16(a0, b1, acc01);
      acc10 = wmma_bf16(a1, b0, acc10);
      acc11 = wmma_bf16(a1, b1, acc11);
    }
    __syncthreads();  // (2) all tileT reads done

    // spill accumulators: C/D layout -> simsBuf[row][col]
    #pragma unroll
    for (int r = 0; r < 8; ++r) {
      int ml = r + 8 * half;
      simsBuf[(wave * 32 + ml)      * 33 + l16     ] = acc00[r];
      simsBuf[(wave * 32 + ml)      * 33 + 16 + l16] = acc01[r];
      simsBuf[(wave * 32 + 16 + ml) * 33 + l16     ] = acc10[r];
      simsBuf[(wave * 32 + 16 + ml) * 33 + 16 + l16] = acc11[r];
    }
    __syncthreads();  // (3) simsBuf visible

    // thread t = query row t: fold strip into running top-3 (no bounds checks:
    // NDB is an exact multiple of 32)
    #pragma unroll 8
    for (int j = 0; j < 32; ++j) {
      top3_update(simsBuf[t * 33 + j], c0 + j, tv0, tv1, tv2, ti0, ti1, ti2);
    }
    // next iteration's barrier (1) orders scan reads vs. future simsBuf writes
  }

  size_t o = ((size_t)t * SIMS_GRID + blockIdx.x) * 3;
  partVal[o + 0] = tv0; partVal[o + 1] = tv1; partVal[o + 2] = tv2;
  partIdx[o + 0] = ti0; partIdx[o + 1] = ti1; partIdx[o + 2] = ti2;
}

// ---------------- 3) merge partial top-3 ----------------
__global__ void merge_topk_kernel(const float* __restrict__ partVal,
                                  const int*   __restrict__ partIdx,
                                  const int*   __restrict__ label,
                                  float* __restrict__ scores,
                                  int*   __restrict__ topIdx,
                                  int*   __restrict__ topLab) {
  __shared__ float lv[256 * 3];
  __shared__ int   li[256 * 3];
  const int row = blockIdx.x, t = threadIdx.x;
  float tv0 = -3.4e38f, tv1 = -3.4e38f, tv2 = -3.4e38f;
  int   ti0 = 0, ti1 = 0, ti2 = 0;
  const size_t base = (size_t)row * SIMS_GRID * 3;
  for (int j = t; j < SIMS_GRID * 3; j += 256) {
    top3_update(partVal[base + j], partIdx[base + j], tv0, tv1, tv2, ti0, ti1, ti2);
  }
  lv[t * 3 + 0] = tv0; lv[t * 3 + 1] = tv1; lv[t * 3 + 2] = tv2;
  li[t * 3 + 0] = ti0; li[t * 3 + 1] = ti1; li[t * 3 + 2] = ti2;
  __syncthreads();
  if (t == 0) {
    float v0 = -3.4e38f, v1 = -3.4e38f, v2 = -3.4e38f;
    int   i0 = 0, i1 = 0, i2 = 0;
    for (int j = 0; j < 256 * 3; ++j)
      top3_update(lv[j], li[j], v0, v1, v2, i0, i1, i2);
    scores[row * 3 + 0] = v0; scores[row * 3 + 1] = v1; scores[row * 3 + 2] = v2;
    topIdx[row * 3 + 0] = i0; topIdx[row * 3 + 1] = i1; topIdx[row * 3 + 2] = i2;
    topLab[row * 3 + 0] = label[i0]; topLab[row * 3 + 1] = label[i1]; topLab[row * 3 + 2] = label[i2];
  }
}

// ---------------- 4) build H = [CLS; query; retrieved] ----------------
__global__ void build_h_kernel(const float* __restrict__ queries,
                               const float* __restrict__ weight,
                               const float* __restrict__ scores,
                               const int*   __restrict__ topIdx,
                               const int*   __restrict__ topLab,
                               float*  __restrict__ H,
                               bf16_t* __restrict__ Hbf) {
  const int seq = blockIdx.x, t = threadIdx.x;
  const int b = seq / TOPK;
  const float sc = scores[seq];
  const int idx = topIdx[seq];
  const int lab = topLab[seq];
  float cls = (t == lab) ? sc : 0.0f;               // one_hot over D, scaled by score
  float hx  = queries[b * DIM + t];
  float hr  = weight[(size_t)t * NDB + idx];        // column idx = normalized db row
  size_t base = (size_t)seq * 3 * DIM;
  H[base + t]           = cls; Hbf[base + t]           = (bf16_t)cls;
  H[base + DIM + t]     = hx;  Hbf[base + DIM + t]     = (bf16_t)hx;
  H[base + 2 * DIM + t] = hr;  Hbf[base + 2 * DIM + t] = (bf16_t)hr;
}

// ---------------- WMMA GEMM: out(M,256) = X(M,256) @ W(256,256) + bias ----------------
// grid.x = 8 (32-col strips), grid.y = M/128, block = 256 (8 waves, 16 rows each).
// Whole 256x32 bf16 W strip staged transposed in LDS once; single barrier.
__global__ __launch_bounds__(256, 1)
void gemm256_kernel(const bf16_t* __restrict__ X,
                    const bf16_t* __restrict__ W,   // bf16, row-major 256x256
                    const float*  __restrict__ bias,
                    float*  __restrict__ outF,
                    bf16_t* __restrict__ outBf,
                    int act) {
  __shared__ __align__(16) bf16_t tileT[32 * TPITCH];
  const int t    = threadIdx.x;
  const int lane = t & 31, wave = t >> 5;
  const int half = lane >> 4, l16 = lane & 15;
  const int mbase = blockIdx.y * 128 + wave * 16;
  const int nbase = blockIdx.x * 32;

  // stage W[0..255][nbase..nbase+31] -> tileT[n][k]; 4 x b128 loads per thread
  {
    const int kq = t >> 2;        // base k row; k = kq + 64*i
    const int n8 = (t & 3) * 8;   // 8-column group
    #pragma unroll
    for (int i = 0; i < 4; ++i) {
      int k = kq + 64 * i;
      bf16x8 v = *(const bf16x8*)(W + (size_t)k * DIM + nbase + n8);
      #pragma unroll
      for (int e = 0; e < 8; ++e) tileT[(n8 + e) * TPITCH + k] = v[e];
    }
  }
  __syncthreads();

  f32x8 acc0 = {}, acc1 = {};
  #pragma unroll
  for (int kb = 0; kb < DIM; kb += 32) {
    bf16x16 b0 = make_frag(&tileT[(0  + l16) * TPITCH + kb + half * 8],
                           &tileT[(0  + l16) * TPITCH + kb + 16 + half * 8]);
    bf16x16 b1 = make_frag(&tileT[(16 + l16) * TPITCH + kb + half * 8],
                           &tileT[(16 + l16) * TPITCH + kb + 16 + half * 8]);
    const bf16_t* ap = X + (size_t)(mbase + l16) * DIM + kb + half * 8;
    bf16x16 a = make_frag(ap, ap + 16);
    acc0 = wmma_bf16(a, b0, acc0);
    acc1 = wmma_bf16(a, b1, acc1);
  }

  #pragma unroll
  for (int r = 0; r < 8; ++r) {
    int m  = mbase + r + 8 * half;
    int c0 = nbase + l16, c1 = nbase + 16 + l16;
    float v0 = acc0[r] + bias[c0];
    float v1 = acc1[r] + bias[c1];
    if (act) { v0 = tanhf(v0); v1 = tanhf(v1); }
    outF[(size_t)m * DIM + c0] = v0;
    outF[(size_t)m * DIM + c1] = v1;
    if (outBf) {
      outBf[(size_t)m * DIM + c0] = (bf16_t)v0;
      outBf[(size_t)m * DIM + c1] = (bf16_t)v1;
    }
  }
}

// ---------------- 5) per-sequence 4-head attention (3 tokens) ----------------
__global__ void attention_kernel(const float* __restrict__ Qf,
                                 const float* __restrict__ Kf,
                                 const float* __restrict__ Vf,
                                 float*  __restrict__ attF,
                                 bf16_t* __restrict__ attBf) {
  __shared__ float Qs[3 * DIM], Ks[3 * DIM], Vs[3 * DIM];
  __shared__ float Sc[4 * 9], Pr[4 * 9];
  const int seq = blockIdx.x, t = threadIdx.x;
  const size_t base = (size_t)seq * 3 * DIM;
  for (int e = t; e < 3 * DIM; e += 256) {
    Qs[e] = Qf[base + e]; Ks[e] = Kf[base + e]; Vs[e] = Vf[base + e];
  }
  __syncthreads();
  if (t < 36) {  // 4 heads x 3x3 scores, dk = 64
    int h = t / 9, r = t % 9, i = r / 3, j = r % 3;
    float s = 0.f;
    for (int d = 0; d < 64; ++d) s += Qs[i * DIM + h * 64 + d] * Ks[j * DIM + h * 64 + d];
    Sc[t] = s * 0.125f;  // 1/sqrt(64)
  }
  __syncthreads();
  if (t < 12) {  // softmax over j
    int h = t / 3, i = t % 3;
    float a0 = Sc[h * 9 + i * 3 + 0], a1 = Sc[h * 9 + i * 3 + 1], a2 = Sc[h * 9 + i * 3 + 2];
    float m = fmaxf(a0, fmaxf(a1, a2));
    float e0 = __expf(a0 - m), e1 = __expf(a1 - m), e2 = __expf(a2 - m);
    float inv = 1.f / (e0 + e1 + e2);
    Pr[h * 9 + i * 3 + 0] = e0 * inv;
    Pr[h * 9 + i * 3 + 1] = e1 * inv;
    Pr[h * 9 + i * 3 + 2] = e2 * inv;
  }
  __syncthreads();
  for (int e = t; e < 3 * DIM; e += 256) {
    int i = e >> 8, d = e & 255, h = d >> 6;
    float o = Pr[h * 9 + i * 3 + 0] * Vs[d] +
              Pr[h * 9 + i * 3 + 1] * Vs[DIM + d] +
              Pr[h * 9 + i * 3 + 2] * Vs[2 * DIM + d];
    attF[base + e] = o;
    attBf[base + e] = (bf16_t)o;
  }
}

// ---------------- 6) gather CLS rows ----------------
__global__ void cls_gather_kernel(const bf16_t* __restrict__ Hbf, bf16_t* __restrict__ clsBf) {
  const int seq = blockIdx.x, t = threadIdx.x;
  clsBf[(size_t)seq * DIM + t] = Hbf[(size_t)seq * 3 * DIM + t];
}

// ---------------- 7) logits = x @ out_w + out_b ----------------
__global__ void logits_kernel(const float* __restrict__ xF,
                              const float* __restrict__ out_w,
                              const float* __restrict__ out_b,
                              float* __restrict__ logitsF) {
  int g = blockIdx.x * 256 + threadIdx.x;
  if (g >= NSEQ * NLAB) return;
  int seq = g / NLAB, l = g % NLAB;
  float s = out_b[l];
  for (int d = 0; d < DIM; ++d) s += xF[(size_t)seq * DIM + d] * out_w[d * NLAB + l];
  logitsF[g] = s;
}

// ---------------- 8) mean over K + retrieval blend ----------------
__global__ void final_kernel(const float* __restrict__ logitsF,
                             const int* __restrict__ topLab,
                             float* __restrict__ out) {
  int g = blockIdx.x * 256 + threadIdx.x;
  if (g >= NQ * NLAB) return;
  int b = g / NLAB, l = g % NLAB;
  float lm = (logitsF[(b * 3 + 0) * NLAB + l] +
              logitsF[(b * 3 + 1) * NLAB + l] +
              logitsF[(b * 3 + 2) * NLAB + l]) * (1.f / 3.f);
  float oh = 0.f;
  for (int k = 0; k < 3; ++k) oh += (topLab[b * 3 + k] == l) ? 1.f : 0.f;
  out[g] = 0.5f * lm + 0.5f * (oh * (1.f / 3.f));
}

// ---------------- host launcher ----------------
extern "C" void kernel_launch(void* const* d_in, const int* in_sizes, int n_in,
                              void* d_out, int out_size, void* d_ws, size_t ws_size,
                              hipStream_t stream) {
  (void)in_sizes; (void)n_in; (void)out_size; (void)ws_size;
  const float* queries = (const float*)d_in[0];
  const float* weight  = (const float*)d_in[1];
  const int*   label   = (const int*)d_in[2];
  const float* Wq = (const float*)d_in[3];
  const float* bq = (const float*)d_in[4];
  const float* Wk = (const float*)d_in[5];
  const float* bk = (const float*)d_in[6];
  const float* Wv = (const float*)d_in[7];
  const float* bv = (const float*)d_in[8];
  const float* Wo = (const float*)d_in[9];
  const float* bo = (const float*)d_in[10];
  const float* dense_w = (const float*)d_in[11];
  const float* dense_b = (const float*)d_in[12];
  const float* out_w   = (const float*)d_in[13];
  const float* out_b   = (const float*)d_in[14];
  float* out = (float*)d_out;

  char* ws = (char*)d_ws;
  size_t off = 0;
  auto alloc = [&](size_t bytes) -> char* {
    off = (off + 255) & ~(size_t)255;
    char* p = ws + off;
    off += bytes;
    return p;
  };
  const size_t WMAT = (size_t)DIM * DIM;  // 65536
  bf16_t* qnBf    = (bf16_t*)alloc((size_t)NQ * DIM * 2);
  bf16_t* wqBf    = (bf16_t*)alloc(NLAYERS_ * WMAT * 2);
  bf16_t* wkBf    = (bf16_t*)alloc(NLAYERS_ * WMAT * 2);
  bf16_t* wvBf    = (bf16_t*)alloc(NLAYERS_ * WMAT * 2);
  bf16_t* woBf    = (bf16_t*)alloc(NLAYERS_ * WMAT * 2);
  bf16_t* denseBf = (bf16_t*)alloc(WMAT * 2);
  float*  partVal = (float*)alloc((size_t)NQ * SIMS_GRID * 3 * 4);
  int*    partIdx = (int*)  alloc((size_t)NQ * SIMS_GRID * 3 * 4);
  float*  scores  = (float*)alloc((size_t)NQ * TOPK * 4);
  int*    topIdx  = (int*)  alloc((size_t)NQ * TOPK * 4);
  int*    topLab  = (int*)  alloc((size_t)NQ * TOPK * 4);
  float*  Hf      = (float*)alloc((size_t)MROWS * DIM * 4);
  bf16_t* Hbf     = (bf16_t*)alloc((size_t)MROWS * DIM * 2);
  float*  Qf      = (float*)alloc((size_t)MROWS * DIM * 4);
  float*  Kf      = (float*)alloc((size_t)MROWS * DIM * 4);
  float*  Vf      = (float*)alloc((size_t)MROWS * DIM * 4);
  float*  attF    = (float*)alloc((size_t)MROWS * DIM * 4);
  bf16_t* attBf   = (bf16_t*)alloc((size_t)MROWS * DIM * 2);
  bf16_t* clsBf   = (bf16_t*)alloc((size_t)NSEQ * DIM * 2);
  float*  xF      = (float*)alloc((size_t)NSEQ * DIM * 4);
  float*  logitsF = (float*)alloc((size_t)NSEQ * NLAB * 4);

  // 1) normalize queries (bf16 A-matrix, L2-resident)
  normalize_kernel<<<NQ, 256, 0, stream>>>(queries, qnBf);

  // 2) pre-convert small weight matrices to bf16
  cvt_kernel<<<(NLAYERS_ * (int)WMAT + 255) / 256, 256, 0, stream>>>(Wq, wqBf, NLAYERS_ * (int)WMAT);
  cvt_kernel<<<(NLAYERS_ * (int)WMAT + 255) / 256, 256, 0, stream>>>(Wk, wkBf, NLAYERS_ * (int)WMAT);
  cvt_kernel<<<(NLAYERS_ * (int)WMAT + 255) / 256, 256, 0, stream>>>(Wv, wvBf, NLAYERS_ * (int)WMAT);
  cvt_kernel<<<(NLAYERS_ * (int)WMAT + 255) / 256, 256, 0, stream>>>(Wo, woBf, NLAYERS_ * (int)WMAT);
  cvt_kernel<<<((int)WMAT + 255) / 256, 256, 0, stream>>>(dense_w, denseBf, (int)WMAT);

  // 3) fused GEMM + top-3 over 500k columns, then merge
  sims_topk_kernel<<<SIMS_GRID, 256, 0, stream>>>(qnBf, weight, partVal, partIdx);
  merge_topk_kernel<<<NQ, 256, 0, stream>>>(partVal, partIdx, label, scores, topIdx, topLab);

  // 4) build H
  build_h_kernel<<<NSEQ, 256, 0, stream>>>(queries, weight, scores, topIdx, topLab, Hf, Hbf);

  // 5) transformer layers
  dim3 gGemm(8, MROWS / 128);  // (8, 18)
  for (int l = 0; l < NLAYERS_; ++l) {
    gemm256_kernel<<<gGemm, 256, 0, stream>>>(Hbf, wqBf + (size_t)l * WMAT, bq + l * DIM, Qf, nullptr, 0);
    gemm256_kernel<<<gGemm, 256, 0, stream>>>(Hbf, wkBf + (size_t)l * WMAT, bk + l * DIM, Kf, nullptr, 0);
    gemm256_kernel<<<gGemm, 256, 0, stream>>>(Hbf, wvBf + (size_t)l * WMAT, bv + l * DIM, Vf, nullptr, 0);
    attention_kernel<<<NSEQ, 256, 0, stream>>>(Qf, Kf, Vf, attF, attBf);
    gemm256_kernel<<<gGemm, 256, 0, stream>>>(attBf, woBf + (size_t)l * WMAT, bo + l * DIM, Hf, Hbf, 0);
  }

  // 6) CLS -> dense(tanh) -> logits -> blend
  cls_gather_kernel<<<NSEQ, 256, 0, stream>>>(Hbf, clsBf);
  dim3 gDense(8, NSEQ / 128);  // (8, 6)
  gemm256_kernel<<<gDense, 256, 0, stream>>>(clsBf, denseBf, dense_b, xF, nullptr, 1);
  logits_kernel<<<(NSEQ * NLAB + 255) / 256, 256, 0, stream>>>(xF, out_w, out_b, logitsF);
  final_kernel<<<(NQ * NLAB + 255) / 256, 256, 0, stream>>>(logitsF, topLab, out);
}
